// ConformerGru_80771154969157
// MI455X (gfx1250) — matrix operand order repaired and checked by reading, not compiled, not run on hardware
//
#include <hip/hip_runtime.h>
#include <math.h>

// ---------------------------------------------------------------------------
// Conformer bidirectional 2-layer GRU + FC head for MI455X (gfx1250).
// Latency-bound recurrence -> batch 16 sequences per workgroup; each GRU step
// is v_wmma_f32_16x16x32_bf16 with f32 accumulation. 3 waves per workgroup,
// one gate (r / z / n) per wave => serial WMMA chain per step is 3x shorter.
// Weights live in LDS pre-swizzled to WMMA B-fragment layout; x tiles are
// double-buffered into LDS with global_load_async_to_lds_b128 (ASYNCcnt);
// gate nonlinearities use CDNA5 v_tanh_f32.
// ---------------------------------------------------------------------------

typedef __attribute__((ext_vector_type(16))) __bf16        v16bf;
typedef __attribute__((ext_vector_type(8)))  float         v8f;
typedef __attribute__((ext_vector_type(8)))  unsigned int  v8u;

union BfCast { v8u u; v16bf b; };

// Hardware packed f32->bf16 conversion (RNE), one VALU op for two values.
static __device__ __forceinline__ unsigned int cvt_pk_bf16(float a, float b) {
  unsigned int r;
  asm("v_cvt_pk_bf16_f32 %0, %1, %2" : "=v"(r) : "v"(a), "v"(b));
  return r;
}
static __device__ __forceinline__ unsigned short f2bf(float f) {
  return (unsigned short)cvt_pk_bf16(f, 0.f);
}
static __device__ __forceinline__ float bf2f(unsigned int s) {
  return __uint_as_float(s << 16);
}
static __device__ __forceinline__ unsigned int load_pair(const float* p) {
  return cvt_pk_bf16(p[0], p[1]);   // f32 source (LDS) -> packed bf16
}
static __device__ __forceinline__ unsigned int load_pair(const unsigned short* p) {
  return *(const unsigned int*)p;   // bf16 source (LDS) -> direct dword
}

// CDNA5 transcendental tanh (co-executes with the WMMA pipe); guarded fallback.
static __device__ __forceinline__ float tanh_hw(float x) {
#if __has_builtin(__builtin_amdgcn_tanhf)
  return __builtin_amdgcn_tanhf(x);
#else
  return 1.f - 2.f * __builtin_amdgcn_rcpf(__expf(2.f * x) + 1.f);
#endif
}
static __device__ __forceinline__ float sigmoid_hw(float x) {
#if __has_builtin(__builtin_amdgcn_tanhf)
  return fmaf(0.5f, __builtin_amdgcn_tanhf(0.5f * x), 0.5f);
#else
  return __builtin_amdgcn_rcpf(1.f + __expf(-x));
#endif
}

// Raw LDS (address_space(3)) byte offset of a __shared__ object, for async ops.
static __device__ __forceinline__ unsigned lds_lo(const void* p) {
  return (unsigned)(unsigned long long)(__attribute__((address_space(3))) const char*)p;
}

static __device__ __forceinline__ v16bf lds_frag(const unsigned int* fragBase, int lane) {
  const unsigned int* p = fragBase + lane * 8;
  BfCast c;
#pragma unroll
  for (int i = 0; i < 8; ++i) c.u[i] = p[i];
  return c.b;
}
static __device__ __forceinline__ v8f wmma_bf16(v16bf a, v16bf b, v8f c) {
  return __builtin_amdgcn_wmma_f32_16x16x32_bf16(false, a, false, b, (short)0, c,
                                                 false, false);
}

// A-fragment K index for 16-bit A 16x32 (ISA 7.12.2): VGPR v, lane-half hl.
static __device__ __forceinline__ int kA(int v, int hl) {
  return ((v < 4) ? 2 * v : 8 + 2 * v) + 8 * hl;
}

// ---------------------------------------------------------------------------
// One direction of one GRU layer for 16 sequences, full T=512 scan.
// grid = 88 blocks (dir = blk/44, seq group = blk%44), block = 3 waves (96):
//   wave 0 -> r gate (tiles 0..3), wave 1 -> z gate (tiles 4..7),
//   wave 2 -> n gate (tiles 8..11) + recurrent state + h/output writeback.
// xin : (704, T, CIN)  f32 (layer0) or bf16 (layer1); row = 256 B either way
// out : (704, T, 128)  bf16, this direction writes columns [dir*64, dir*64+64)
// ---------------------------------------------------------------------------
template <int CIN, typename TI>
__global__ __launch_bounds__(96) void gru_scan_kernel(
    const TI* __restrict__ xin, const float* __restrict__ Wih,
    const float* __restrict__ Whh, const float* __restrict__ bih,
    const float* __restrict__ bhh, unsigned short* __restrict__ out) {
  constexpr int T = 512;
  constexpr int KFX = CIN / 32;                    // x-proj K fragments per tile
  constexpr int ROWB = CIN * (int)sizeof(TI);      // 256 bytes for both layers
  const int tid = threadIdx.x;
  const int lane = tid & 31;
  const int wid = tid >> 5;                        // 0=r, 1=z, 2=n
  const int dir = blockIdx.x / 44;
  const int seq0 = (blockIdx.x % 44) * 16;
  const int m16 = lane & 15;
  const int hl = lane >> 4;

  // Weight fragments pre-swizzled to WMMA B layout: [tile][kf][lane][v] dwords.
  __shared__ unsigned int wihF[12 * KFX * 32 * 8];
  __shared__ unsigned int whhF[12 * 2 * 32 * 8];
  __shared__ __align__(4)  unsigned short hbuf[2][16 * 64];
  __shared__ __align__(16) unsigned char  xstage[2][16 * 256];  // 16 rows x 256B
  __shared__ float gatebuf[2][16 * 64];            // staged sigmoid(r), sigmoid(z)

  const float* WihD = Wih + (size_t)dir * 192 * CIN;
  const float* WhhD = Whh + (size_t)dir * 192 * 64;

  // B-fragment element (lane l, vgpr v): N = l%16, K = (v<4 ? 2v : 8+2v) + 8*(l>=16).
  for (int e = tid; e < 12 * 2 * 32 * 8; e += 96) {
    int v = e & 7, ln = (e >> 3) & 31, kf = (e >> 8) & 1, tile = e >> 9;
    int j = 16 * tile + (ln & 15);
    int k = 32 * kf + kA(v, ln >> 4);
    whhF[e] = cvt_pk_bf16(WhhD[j * 64 + k], WhhD[j * 64 + k + 1]);
  }
  for (int e = tid; e < 12 * KFX * 32 * 8; e += 96) {
    int v = e & 7, ln = (e >> 3) & 31, r = e >> 8, kf = r % KFX, tile = r / KFX;
    int j = 16 * tile + (ln & 15);
    int k = 32 * kf + kA(v, ln >> 4);
    wihF[e] = cvt_pk_bf16(WihD[j * CIN + k], WihD[j * CIN + k + 1]);
  }
  for (int i = tid; i < 16 * 64 / 2; i += 96) ((unsigned int*)hbuf[0])[i] = 0u;
  __syncthreads();

  // Async-stage one timestep tile (16 rows x 256 B) into xstage[buf]; wave 0 only.
  const unsigned xbase = lds_lo(&xstage[0][0]);
  auto stage = [&](int txs, int buf) {
#pragma unroll
    for (int i = 0; i < 8; ++i) {
      int c = i * 32 + lane;                       // 256 chunks of 16 B
      int row = c >> 4, cib = (c & 15) * 16;
      const unsigned char* gp = (const unsigned char*)xin +
          ((size_t)(seq0 + row) * T + txs) * ROWB + cib;
      unsigned ldsOff = xbase + (unsigned)(buf * 4096 + c * 16);
      asm volatile("global_load_async_to_lds_b128 %0, %1, off"
                   :: "v"(ldsOff), "v"((unsigned long long)gp) : "memory");
    }
  };
  if (wid == 0) stage(dir ? T - 1 : 0, 0);

  // Per-wave bias, folded into accumulator init (depends only on N column).
  // wid<2: biasA = bih+bhh for that gate. wid==2: biasA = bih_n, biasB = bhh_n.
  float biasA[4], biasB[4];
#pragma unroll
  for (int tn = 0; tn < 4; ++tn) {
    int g = dir * 192 + 64 * wid + 16 * tn + m16;
    if (wid < 2) { biasA[tn] = bih[g] + bhh[g]; biasB[tn] = 0.f; }
    else         { biasA[tn] = bih[g];          biasB[tn] = bhh[g]; }
  }

  // Hidden state (wave 2 only) in D-fragment layout:
  // hD[tn][v] = (M = v + 8*hl, k = 16*tn + m16).
  float hD[4][8];
#pragma unroll
  for (int tn = 0; tn < 4; ++tn)
#pragma unroll
    for (int j = 0; j < 8; ++j) hD[tn][j] = 0.f;

  int cur = 0;
  for (int t = 0; t < T; ++t) {
    const int tx = dir ? (T - 1 - t) : t;
    const int xb = t & 1;

    if (wid == 0)  // ASYNCcnt is per-wave; issuer waits, then B1 publishes tile
      asm volatile("s_wait_asynccnt 0" ::: "memory");
    __syncthreads();  // B1: x tile for step t + hbuf[cur] visible to all waves

    // ---- A fragments (built redundantly by each wave) ----
    v16bf aH[2];
#pragma unroll
    for (int kf = 0; kf < 2; ++kf) {
      BfCast c;
#pragma unroll
      for (int v = 0; v < 8; ++v) {
        int k = 32 * kf + kA(v, hl);
        c.u[v] = *(const unsigned int*)&hbuf[cur][m16 * 64 + k];
      }
      aH[kf] = c.b;
    }
    const TI* xr = (const TI*)&xstage[xb][m16 * 256];
    v16bf aX[KFX];
#pragma unroll
    for (int kf = 0; kf < KFX; ++kf) {
      BfCast c;
#pragma unroll
      for (int v = 0; v < 8; ++v) {
        int k = 32 * kf + kA(v, hl);
        c.u[v] = load_pair(xr + k);
      }
      aX[kf] = c.b;
    }

    // Kick off next timestep's async copy; overlaps the WMMA chains below.
    if (wid == 0 && t + 1 < T) stage(dir ? tx - 1 : tx + 1, xb ^ 1);

    // ---- per-gate WMMA chains ----
    v8f cx[4], ch[4];  // n-gate accumulators (live across B2 on wave 2)
    if (wid < 2) {
#pragma unroll
      for (int tn = 0; tn < 4; ++tn) {
        const int tile = 4 * wid + tn;
        v8f c;
#pragma unroll
        for (int j = 0; j < 8; ++j) c[j] = biasA[tn];
#pragma unroll
        for (int kf = 0; kf < KFX; ++kf)
          c = wmma_bf16(aX[kf], lds_frag(wihF + (tile * KFX + kf) * 256, lane), c);
#pragma unroll
        for (int kf = 0; kf < 2; ++kf)
          c = wmma_bf16(aH[kf], lds_frag(whhF + (tile * 2 + kf) * 256, lane), c);
        const int kcol = 16 * tn + m16;
#pragma unroll
        for (int j = 0; j < 8; ++j)
          gatebuf[wid][(j + 8 * hl) * 64 + kcol] = sigmoid_hw(c[j]);
      }
    } else {
#pragma unroll
      for (int tn = 0; tn < 4; ++tn) {
        const int tile = 8 + tn;
#pragma unroll
        for (int j = 0; j < 8; ++j) { cx[tn][j] = biasA[tn]; ch[tn][j] = biasB[tn]; }
#pragma unroll
        for (int kf = 0; kf < KFX; ++kf)
          cx[tn] = wmma_bf16(aX[kf], lds_frag(wihF + (tile * KFX + kf) * 256, lane), cx[tn]);
#pragma unroll
        for (int kf = 0; kf < 2; ++kf)
          ch[tn] = wmma_bf16(aH[kf], lds_frag(whhF + (tile * 2 + kf) * 256, lane), ch[tn]);
      }
    }
    __syncthreads();  // B2: sigmoid(r), sigmoid(z) staged in LDS

    // ---- n-wave: combine gates, update state, write h and output ----
    if (wid == 2) {
#pragma unroll
      for (int tn = 0; tn < 4; ++tn) {
        const int kcol = 16 * tn + m16;
#pragma unroll
        for (int j = 0; j < 8; ++j) {
          const int m = j + 8 * hl;
          float r = gatebuf[0][m * 64 + kcol];
          float z = gatebuf[1][m * 64 + kcol];
          float nv = tanh_hw(cx[tn][j] + r * ch[tn][j]);
          float hnew = fmaf(z, hD[tn][j] - nv, nv);  // (1-z)*n + z*h
          hD[tn][j] = hnew;
          unsigned short hb = f2bf(hnew);
          hbuf[cur ^ 1][m * 64 + kcol] = hb;
          out[((size_t)(seq0 + m) * T + tx) * 128 + dir * 64 + kcol] = hb;
        }
      }
    }
    cur ^= 1;
    // B1 of the next iteration publishes hbuf[cur] before anyone reads it.
  }
}

// ---------------------------------------------------------------------------
// FC(128 -> 1) + sigmoid + (b*f, t) -> (b, t, f) transpose.
// ---------------------------------------------------------------------------
__global__ __launch_bounds__(256) void fc_sigmoid_kernel(
    const unsigned short* __restrict__ h, const float* __restrict__ w,
    const float* __restrict__ bias, float* __restrict__ outp) {
  int idx = blockIdx.x * blockDim.x + threadIdx.x;
  if (idx >= 704 * 512) return;
  int n = idx >> 9;  // sequence (b*88 + f)
  int t = idx & 511;
  const unsigned short* row = h + (size_t)idx * 128;
  float acc = bias[0];
#pragma unroll 8
  for (int c = 0; c < 128; c += 2) {
    unsigned int pr = *(const unsigned int*)(row + c);
    acc += bf2f(pr & 0xffffu) * w[c] + bf2f(pr >> 16) * w[c + 1];
  }
  int b = n / 88, f = n % 88;
  outp[((size_t)b * 512 + t) * 88 + f] = sigmoid_hw(acc);
}

// ---------------------------------------------------------------------------
extern "C" void kernel_launch(void* const* d_in, const int* in_sizes, int n_in,
                              void* d_out, int out_size, void* d_ws, size_t ws_size,
                              hipStream_t stream) {
  (void)in_sizes; (void)n_in; (void)out_size; (void)ws_size;
  const float* x    = (const float*)d_in[0];   // (8,88,512,64) == (704,512,64)
  const float* Wih0 = (const float*)d_in[1];   // (2,192,64)
  const float* Whh0 = (const float*)d_in[2];   // (2,192,64)
  const float* bih0 = (const float*)d_in[3];   // (2,192)
  const float* bhh0 = (const float*)d_in[4];   // (2,192)
  const float* Wih1 = (const float*)d_in[5];   // (2,192,128)
  const float* Whh1 = (const float*)d_in[6];   // (2,192,64)
  const float* bih1 = (const float*)d_in[7];   // (2,192)
  const float* bhh1 = (const float*)d_in[8];   // (2,192)
  const float* fc_w = (const float*)d_in[9];   // (1,128)
  const float* fc_b = (const float*)d_in[10];  // (1,)

  // Workspace: two (704,512,128) bf16 inter-layer buffers (~92 MB each).
  unsigned short* buf0 = (unsigned short*)d_ws;
  unsigned short* buf1 = buf0 + (size_t)704 * 512 * 128;

  gru_scan_kernel<64, float><<<dim3(88), dim3(96), 0, stream>>>(
      x, Wih0, Whh0, bih0, bhh0, buf0);
  gru_scan_kernel<128, unsigned short><<<dim3(88), dim3(96), 0, stream>>>(
      buf0, Wih1, Whh1, bih1, bhh1, buf1);
  fc_sigmoid_kernel<<<dim3((704 * 512 + 255) / 256), dim3(256), 0, stream>>>(
      buf1, fc_w, fc_b, (float*)d_out);
}
